// LittleBitLinear_24412594110603
// MI455X (gfx1250) — matrix operand order repaired and checked by reading, not compiled, not run on hardware
//
#include <hip/hip_runtime.h>
#include <hip/hip_bf16.h>

// ---------------------------------------------------------------------------
// LittleBitLinear for MI455X (gfx1250):
//   M = (sign(U) * ell) @ sign(V)^T, scaled by g (cols) and h (rows)
// Strategy:
//   Pass 1: pack A[o,r] = sign(U[o,r]) * ell[r]  -> fp16   (8 MB in d_ws)
//           pack B[i,r] = sign(V[i,r])           -> fp16   (22.5 MB in d_ws)
//           (U/V read with non-temporal loads: single-use streams;
//            packed A/B stored regular-temporal so they stay in the 192MB L2)
//   Pass 2: tiled WMMA GEMM, f32 accumulate; epilogue applies h[o]*g[i] in
//           f32 and stores NON-TEMPORAL (output is write-once, 180 MB; keep
//           it from rinsing the L2-resident operands).
// Roofline: 92.3 GFLOP f16-WMMA vs ~180 MB HBM writes @ 23.3 TB/s -> both in
// the tens-of-us regime; matrix engine + L2 residency is the winning path.
// ---------------------------------------------------------------------------

#define D_OUT 4096
#define D_IN  11008
#define RDIM  1024

typedef __attribute__((ext_vector_type(16))) _Float16 v16h;
typedef __attribute__((ext_vector_type(8)))  _Float16 v8h;
typedef __attribute__((ext_vector_type(4)))  _Float16 v4h;
typedef __attribute__((ext_vector_type(8)))  float    v8f;
typedef __attribute__((ext_vector_type(4)))  float    v4f;   // builtin-compatible float4

// ---------------------------------------------------------------------------
// Pass 1a: A[o,r] = (U[o,r] >= 0 ? +ell[r] : -ell[r]) as fp16
// ---------------------------------------------------------------------------
__global__ __launch_bounds__(256) void lbl_pack_A(const float* __restrict__ U,
                                                  const float* __restrict__ ell,
                                                  _Float16* __restrict__ A) {
    const int idx = (blockIdx.x * blockDim.x + threadIdx.x) * 4;
    if (idx >= D_OUT * RDIM) return;
    const int r = idx & (RDIM - 1);           // RDIM power of two
    const v4f u = __builtin_nontemporal_load((const v4f*)(U + idx));
    const v4f e = *(const v4f*)(ell + r);
    v4h a;
#pragma unroll
    for (int j = 0; j < 4; ++j)
        a[j] = (_Float16)(u[j] >= 0.0f ? e[j] : -e[j]);
    *(v4h*)(A + idx) = a;                     // regular-temporal: keep in L2
}

// ---------------------------------------------------------------------------
// Pass 1b: B[i,r] = sign(V[i,r]) in {+1,-1} as fp16 (exact)
// ---------------------------------------------------------------------------
__global__ __launch_bounds__(256) void lbl_pack_B(const float* __restrict__ V,
                                                  _Float16* __restrict__ B) {
    const int idx = (blockIdx.x * blockDim.x + threadIdx.x) * 4;
    if (idx >= D_IN * RDIM) return;
    const v4f v = __builtin_nontemporal_load((const v4f*)(V + idx));
    v4h b;
#pragma unroll
    for (int j = 0; j < 4; ++j)
        b[j] = (_Float16)(v[j] >= 0.0f ? 1.0f : -1.0f);
    *(v4h*)(B + idx) = b;                     // regular-temporal: keep in L2
}

// ---------------------------------------------------------------------------
// Pass 2: WMMA GEMM.
//   out[o,i] = h[o] * g[i] * sum_r A[o,r] * B[i,r]
// Block = 256 threads = 8 waves (2x4); each wave owns a 64x64 output tile =
// 4x4 grid of v_wmma_f32_16x16x32_f16. WG tile = 128x256.
// Grid = (11008/256, 4096/128) = (43, 32) -> exact, no bounds checks.
//
// Fragment layouts per CDNA5 ISA 7.12.2 (wave32):
//  A (16x32 f16): lane L<16  -> row L,    elems = K[+0..7]  ++ K[+16..23]
//                 lane L>=16 -> row L-16, elems = K[+8..15] ++ K[+24..31]
//  B (32x16 f16): lane n<16  -> col n,    elems = K[+0..7]  ++ K[+16..23]
//                 lane n>=16 -> col n-16, elems = K[+8..15] ++ K[+24..31]
//  C/D (16x16 f32): VGPR v -> row = v + 8*(lane>=16), col = lane%16
// All per-lane pieces are contiguous 16B runs -> aligned global_load_b128.
//
// K loop unrolled x4 (64 WMMAs per macro-iter) so the per-iteration pointer
// updates + WMMA->VALU hazard NOPs amortize; inner K offsets fold into the
// 24-bit instruction offset. WGP-scope speculative prefetch of the next
// macro-chunk is issued unconditionally (OOB speculative prefetch is
// silently dropped per ISA), avoiding a guard branch.
// ---------------------------------------------------------------------------
__global__ __launch_bounds__(256) void lbl_wmma_gemm(const _Float16* __restrict__ A,
                                                     const _Float16* __restrict__ B,
                                                     const float* __restrict__ h,
                                                     const float* __restrict__ g,
                                                     float* __restrict__ out) {
    const int lane  = threadIdx.x & 31;
    const int wave  = threadIdx.x >> 5;
    const int lmod  = lane & 15;
    const int lhalf = lane >> 4;                 // 0: lanes 0-15, 1: lanes 16-31

    const int rowBase = blockIdx.y * 128 + (wave >> 2) * 64;   // along D_OUT
    const int colBase = blockIdx.x * 256 + (wave & 3) * 64;    // along D_IN

    // Per-lane base pointers into the packed fp16 buffers.
    const _Float16* aPtr[4];
    const _Float16* bPtr[4];
#pragma unroll
    for (int mi = 0; mi < 4; ++mi)
        aPtr[mi] = A + (size_t)(rowBase + mi * 16 + lmod) * RDIM + lhalf * 8;
#pragma unroll
    for (int ni = 0; ni < 4; ++ni)
        bPtr[ni] = B + (size_t)(colBase + ni * 16 + lmod) * RDIM + lhalf * 8;

    v8f acc[4][4];
#pragma unroll
    for (int mi = 0; mi < 4; ++mi)
#pragma unroll
        for (int ni = 0; ni < 4; ++ni)
            acc[mi][ni] = (v8f){0.f, 0.f, 0.f, 0.f, 0.f, 0.f, 0.f, 0.f};

    for (int k0 = 0; k0 < RDIM; k0 += 128) {     // macro-iter: 4 K-chunks
        // Speculative WGP-scope prefetch of the next macro-chunk (256B/row).
        // Past-the-end prefetch on the final iteration is silently dropped.
#pragma unroll
        for (int mi = 0; mi < 4; ++mi)
            __builtin_prefetch(aPtr[mi] + k0 + 128, 0, 3);
#pragma unroll
        for (int ni = 0; ni < 4; ++ni)
            __builtin_prefetch(bPtr[ni] + k0 + 128, 0, 3);

#pragma unroll
        for (int kk = 0; kk < 128; kk += 32) {
            const int k = k0 + kk;
            v16h afrag[4], bfrag[4];

#pragma unroll
            for (int mi = 0; mi < 4; ++mi) {
                const v8h lo = *(const v8h*)(aPtr[mi] + k);       // K chunk 0
                const v8h hi = *(const v8h*)(aPtr[mi] + k + 16);  // K chunk +16
#pragma unroll
                for (int j = 0; j < 8; ++j) {
                    afrag[mi][j]     = lo[j];
                    afrag[mi][8 + j] = hi[j];
                }
            }
#pragma unroll
            for (int ni = 0; ni < 4; ++ni) {
                const v8h lo = *(const v8h*)(bPtr[ni] + k);       // K chunk 0
                const v8h hi = *(const v8h*)(bPtr[ni] + k + 16);  // K chunk +16
#pragma unroll
                for (int j = 0; j < 8; ++j) {
                    bfrag[ni][j]     = lo[j];
                    bfrag[ni][8 + j] = hi[j];
                }
            }

#pragma unroll
            for (int mi = 0; mi < 4; ++mi)
#pragma unroll
                for (int ni = 0; ni < 4; ++ni)
                    acc[mi][ni] = __builtin_amdgcn_wmma_f32_16x16x32_f16(
                        /*neg_a=*/false, afrag[mi],
                        /*neg_b=*/false, bfrag[ni],
                        /*c_mod=*/(short)0, acc[mi][ni],
                        /*reuse_a=*/false, /*reuse_b=*/false);
        }
    }

    // Epilogue: apply h (row) and g (col) in f32; non-temporal stores so the
    // write-once 180MB output doesn't evict the L2-resident A/B operands.
#pragma unroll
    for (int mi = 0; mi < 4; ++mi) {
        float hr[8];
#pragma unroll
        for (int v = 0; v < 8; ++v)
            hr[v] = h[rowBase + mi * 16 + lhalf * 8 + v];
#pragma unroll
        for (int ni = 0; ni < 4; ++ni) {
            const int col = colBase + ni * 16 + lmod;
            const float gc = g[col];
#pragma unroll
            for (int v = 0; v < 8; ++v) {
                const int row = rowBase + mi * 16 + lhalf * 8 + v;
                __builtin_nontemporal_store(acc[mi][ni][v] * gc * hr[v],
                                            out + (size_t)row * D_IN + col);
            }
        }
    }
}

// ---------------------------------------------------------------------------
// Launch: inputs in setup_inputs() order: U_fp, V_fp, h, g, ell
// Workspace layout: [ Apack: 4096*1024 fp16 | Bpack: 11008*1024 fp16 ] ~29.5MB
// ---------------------------------------------------------------------------
extern "C" void kernel_launch(void* const* d_in, const int* in_sizes, int n_in,
                              void* d_out, int out_size, void* d_ws, size_t ws_size,
                              hipStream_t stream) {
    const float* U   = (const float*)d_in[0];
    const float* V   = (const float*)d_in[1];
    const float* h   = (const float*)d_in[2];
    const float* g   = (const float*)d_in[3];
    const float* ell = (const float*)d_in[4];
    float* out = (float*)d_out;

    _Float16* Apack = (_Float16*)d_ws;
    _Float16* Bpack = Apack + (size_t)D_OUT * RDIM;

    {   // Pass 1a: pack A (4 elems/thread)
        const int total  = D_OUT * RDIM / 4;
        const int blocks = (total + 255) / 256;
        lbl_pack_A<<<blocks, 256, 0, stream>>>(U, ell, Apack);
    }
    {   // Pass 1b: pack B (4 elems/thread)
        const int total  = D_IN * RDIM / 4;
        const int blocks = (total + 255) / 256;
        lbl_pack_B<<<blocks, 256, 0, stream>>>(V, Bpack);
    }
    {   // Pass 2: WMMA GEMM, WG tile 128x256
        dim3 grid(D_IN / 256, D_OUT / 128);    // (43, 32)
        lbl_wmma_gemm<<<grid, 256, 0, stream>>>(Apack, Bpack, h, g, out);
    }
}
